// GridAttention_63153199120659
// MI455X (gfx1250) — matrix operand order repaired and checked by reading, not compiled, not run on hardware
//
#include <hip/hip_runtime.h>
#include <hip/hip_bf16.h>

// ---------------------------------------------------------------------------
// GridAttention, algebraically refactored to be HBM-bound on MI455X.
//
// Shapes: B=8, D=768, W=128, nt=8 -> gs=16, T=64 tiles, S=256, h=12, hd=64.
// kv[b,t,s,dd] = x[b, 3s + dd/256, ty*16 + (dd%256)/16, tx*16 + dd%16]
// scores[h,s]  = g_h . kv_s          (g_h = (q Wq_h^T + bq_h) Wk_h / 8;
//                                     the r.bk constant cancels in softmax)
// ctx_h        = (sum_s attn kv_s) Wv_h^T + bv_h
//
// kv sub-tiles (128 s x 128 dd fp32 = 64 KB) are staged by the Tensor Data
// Mover into a double-buffered LDS pipeline (issue-ahead-by-2, in-order
// TENSORcnt), so the DMA of stage i+1 overlaps the WMMA of stage i.
// ---------------------------------------------------------------------------

typedef __attribute__((ext_vector_type(16))) _Float16 v16h;
typedef __attribute__((ext_vector_type(8)))  _Float16 v8h;
typedef __attribute__((ext_vector_type(4)))  _Float16 v4h;
typedef __attribute__((ext_vector_type(8)))  float    v8f;
typedef __attribute__((ext_vector_type(4)))  float    v4f;
typedef __attribute__((ext_vector_type(4)))  unsigned int v4u;
typedef __attribute__((ext_vector_type(8)))  int      v8i;
typedef __attribute__((ext_vector_type(4)))  int      v4i;

#define D_CH      768
#define CH_STRIDE 16384   // 128*128
#define NTILE     512     // B*T

static __device__ __forceinline__ v16h cat16(v8h lo, v8h hi) {
  return __builtin_shufflevector(lo, hi, 0,1,2,3,4,5,6,7,8,9,10,11,12,13,14,15);
}
static __device__ __forceinline__ v8f wmma32(v16h a, v16h b, v8f c) {
  return __builtin_amdgcn_wmma_f32_16x16x32_f16(false, a, false, b, (short)0, c,
                                                false, false);
}
static __device__ __forceinline__ float wave_max(float v) {
  #pragma unroll
  for (int m = 16; m > 0; m >>= 1) v = fmaxf(v, __shfl_xor(v, m, 32));
  return v;
}
static __device__ __forceinline__ float wave_sum(float v) {
  #pragma unroll
  for (int m = 16; m > 0; m >>= 1) v += __shfl_xor(v, m, 32);
  return v;
}

// TDM: stage one 128s x 128dd fp32 kv sub-tile (64 KB):
//   x run: 16 contiguous floats (gx), y: 8 rows stride 128 (gy half),
//   z: 128 steps stride 49152 (3 channels per s).  D# per ISA §8.3-8.6.
static __device__ __forceinline__ void tdm_stage_kv(const float* gsrc,
                                                    float* ldsdst) {
  const unsigned long long ga = (unsigned long long)(uintptr_t)gsrc;  // bytes
  const unsigned int la = (unsigned int)(uintptr_t)ldsdst;            // LDS off
  v4u g0;
  g0[0] = 1u;                                            // count=1, user D#
  g0[1] = la;                                            // lds_addr
  g0[2] = (unsigned int)ga;                              // global_addr[31:0]
  g0[3] = (unsigned int)((ga >> 32) & 0x01FFFFFFu) | (2u << 30);  // [56:32]|type=2
  v8i g1;
  g1[0] = (int)(2u << 16);            // data_size = 4 bytes; mask/flags = 0
  g1[1] = (int)(16u << 16);           // tensor_dim0 = 16
  g1[2] = (int)(8u << 16);            // tensor_dim1 = 8
  g1[3] = (int)(16u << 16);           // tile_dim0 = 16
  g1[4] = (int)(8u | (128u << 16));   // tile_dim1 = 8, tile_dim2 = 128
  g1[5] = 128;                        // tensor_dim0_stride = 128 elems
  g1[6] = (int)(49152u << 16);        // tensor_dim1_stride[15:0] = 0xC000
  g1[7] = 0;                          // tensor_dim1_stride[47:16] = 0
  v4i g2; g2[0] = 128; g2[1] = 0; g2[2] = 0; g2[3] = 0;  // tensor_dim2 = 128
  v4i g3; g3[0] = 0;  g3[1] = 0;  g3[2] = 0;  g3[3] = 0;
  v8i g4; g4[0] = 0; g4[1] = 0; g4[2] = 0; g4[3] = 0;    // unused trailing slot
  g4[4] = 0; g4[5] = 0; g4[6] = 0; g4[7] = 0;
  __builtin_amdgcn_tensor_load_to_lds(g0, g1, g2, g3, g4, 0);
}

// ---------------------------------------------------------------------------
// Kernel 1: tile-mean pooling.  Qmean[bt][d] = mean over 16x16 patch.
// x is streamed once (nontemporal: it cannot stay in L2 until attn re-reads).
// ---------------------------------------------------------------------------
__global__ __launch_bounds__(256) void pool_kernel(const float* __restrict__ x,
                                                   float* __restrict__ Qmean) {
  const int bt = blockIdx.x;
  const int b = bt >> 6, tt = bt & 63, ty = tt >> 3, tx = tt & 7;
  const int t = threadIdx.x, lane = t & 31, w = t >> 5;
  const float* xb = x + ((size_t)b * D_CH) * CH_STRIDE + (ty * 16) * 128 + tx * 16;
  const int sp0 = lane * 4;
  const int off0 = (sp0 >> 4) * 128 + (sp0 & 15);
  const int sp1 = 128 + lane * 4;
  const int off1 = (sp1 >> 4) * 128 + (sp1 & 15);
  for (int d = w; d < D_CH; d += 16) {          // two channels in flight
    const float* xc0 = xb + (size_t)d * CH_STRIDE;
    const float* xc1 = xb + (size_t)(d + 8) * CH_STRIDE;
    v4f a0 = __builtin_nontemporal_load((const v4f*)(xc0 + off0));
    v4f a1 = __builtin_nontemporal_load((const v4f*)(xc0 + off1));
    v4f b0 = __builtin_nontemporal_load((const v4f*)(xc1 + off0));
    v4f b1 = __builtin_nontemporal_load((const v4f*)(xc1 + off1));
    float sa = a0[0]+a0[1]+a0[2]+a0[3] + a1[0]+a1[1]+a1[2]+a1[3];
    float sb = b0[0]+b0[1]+b0[2]+b0[3] + b1[0]+b1[1]+b1[2]+b1[3];
    sa = wave_sum(sa);
    sb = wave_sum(sb);
    if (lane == 0) {
      Qmean[(size_t)bt * D_CH + d]     = sa * (1.f / 256.f);
      Qmean[(size_t)bt * D_CH + d + 8] = sb * (1.f / 256.f);
    }
  }
}

// ---------------------------------------------------------------------------
// Generic batched f16-WMMA GEMM:  C = alpha * A @ B' (+ bias[n])
//   A (MxK) row-major fp32, lda.   B' = transB ? B[n*ldb+k] : B[k*ldb+n].
//   Block tile 128x64, 8 waves (4x2), each wave 32x32 (4 WMMAs / k-step).
//   M % 128 == 0, N % 64 == 0, K % 32 == 0 at all call sites.
// ---------------------------------------------------------------------------
__global__ __launch_bounds__(256) void gemm_f16w(
    const float* __restrict__ A, const float* __restrict__ B,
    const float* __restrict__ bias, float* __restrict__ C,
    int M, int N, int K, int lda, int ldb, int ldc,
    int transB, float alpha,
    long long sA, long long sB, long long sC, long long sBias) {
  const int bz = blockIdx.z;
  A += sA * bz; B += sB * bz; C += sC * bz;
  if (bias) bias += sBias * bz;

  const int n0 = blockIdx.x * 64, m0 = blockIdx.y * 128;
  __shared__ _Float16 Al[128][40];   // padded rows: 80B stride, 16B aligned
  __shared__ _Float16 Bl[64][40];

  const int t = threadIdx.x, lane = t & 31, w = t >> 5;
  const int wm = w & 3, wn = w >> 2;
  const int half = lane >> 4, l16 = lane & 15;

  v8f acc[2][2] = {};

  for (int k0 = 0; k0 < K; k0 += 32) {
    __syncthreads();
    // ---- A panel 128x32: 4 v4f loads in flight, then stores ----
    v4f at[4];
    #pragma unroll
    for (int u = 0; u < 4; ++u) {
      int idx = u * 256 + t;
      int am = idx >> 3, ak4 = (idx & 7) * 4;
      at[u] = *(const v4f*)&A[(long long)(m0 + am) * lda + (k0 + ak4)];
    }
    #pragma unroll
    for (int u = 0; u < 4; ++u) {
      int idx = u * 256 + t;
      int am = idx >> 3, ak4 = (idx & 7) * 4;
      v4h hv;
      #pragma unroll
      for (int e = 0; e < 4; ++e) hv[e] = (_Float16)at[u][e];
      *(v4h*)&Al[am][ak4] = hv;
    }
    // ---- B panel 32x64 -> Bl[n][k] ----
    if (transB) {
      v4f btv[2];
      #pragma unroll
      for (int u = 0; u < 2; ++u) {
        int idx = u * 256 + t;
        int bn = idx >> 3, bk4 = (idx & 7) * 4;
        btv[u] = *(const v4f*)&B[(long long)(n0 + bn) * ldb + (k0 + bk4)];
      }
      #pragma unroll
      for (int u = 0; u < 2; ++u) {
        int idx = u * 256 + t;
        int bn = idx >> 3, bk4 = (idx & 7) * 4;
        v4h hv;
        #pragma unroll
        for (int e = 0; e < 4; ++e) hv[e] = (_Float16)btv[u][e];
        *(v4h*)&Bl[bn][bk4] = hv;
      }
    } else {
      v4f btv[2];
      #pragma unroll
      for (int u = 0; u < 2; ++u) {
        int idx = u * 256 + t;
        int bk = idx >> 4, bn4 = (idx & 15) * 4;
        btv[u] = *(const v4f*)&B[(long long)(k0 + bk) * ldb + (n0 + bn4)];
      }
      #pragma unroll
      for (int u = 0; u < 2; ++u) {
        int idx = u * 256 + t;
        int bk = idx >> 4, bn4 = (idx & 15) * 4;
        #pragma unroll
        for (int e = 0; e < 4; ++e) Bl[bn4 + e][bk] = (_Float16)btv[u][e];
      }
    }
    __syncthreads();

    v16h af[2], bf[2];
    #pragma unroll
    for (int i = 0; i < 2; ++i) {                  // A frag: 16x32 layout
      int am = wm * 32 + i * 16 + l16;
      v8h lo = *(const v8h*)&Al[am][half * 8];
      v8h hi = *(const v8h*)&Al[am][16 + half * 8];
      af[i] = cat16(lo, hi);
    }
    #pragma unroll
    for (int j = 0; j < 2; ++j) {                  // B frag: 32x16 layout
      int bn = wn * 32 + j * 16 + l16;
      v8h lo = *(const v8h*)&Bl[bn][half * 8];
      v8h hi = *(const v8h*)&Bl[bn][16 + half * 8];
      bf[j] = cat16(lo, hi);
    }
    #pragma unroll
    for (int i = 0; i < 2; ++i)
      #pragma unroll
      for (int j = 0; j < 2; ++j)
        acc[i][j] = wmma32(af[i], bf[j], acc[i][j]);
  }

  #pragma unroll
  for (int i = 0; i < 2; ++i)
    #pragma unroll
    for (int j = 0; j < 2; ++j)
      #pragma unroll
      for (int v = 0; v < 8; ++v) {
        int row = m0 + wm * 32 + i * 16 + v + 8 * half;
        int col = n0 + wn * 32 + j * 16 + l16;
        float out = alpha * acc[i][j][v];
        if (bias) out += bias[col];
        C[(long long)row * ldc + col] = out;
      }
}

// ---------------------------------------------------------------------------
// Kernel 3: per-tile attention.  One block per (b,t), 8 waves.
//   Double-buffered TDM pipeline: 12 stages per pass, issue-ahead-by-2.
//   scores (16h x 256s, K=768) and cbar (16h x 768dd, K=256) via WMMA.
// LDS: kv 2x64KB(f32) + gthird 8KB + slds 16KB + attnT 8KB = 160KB.
// ---------------------------------------------------------------------------
__global__ __launch_bounds__(256) void attn_kernel(const float* __restrict__ x,
                                                   const float* __restrict__ G,
                                                   float* __restrict__ Cbar) {
  const int bt = blockIdx.x;
  const int b = bt >> 6, tt = bt & 63, ty = tt >> 3, tx = tt & 7;
  const int t = threadIdx.x, lane = t & 31, w = t >> 5;
  const int half = lane >> 4, l16 = lane & 15;

  __shared__ float    kvlds[2][128][128]; // double-buffered sub-tiles, fp32
  __shared__ _Float16 gthird[16][256];    // g rows (12 used, 4 zero)
  __shared__ float    slds[16][256];      // scores; later cbar staging
  __shared__ _Float16 attnT[16][256];     // softmax(scores)

  const float* xb = x + ((size_t)b * D_CH) * CH_STRIDE + (ty * 16) * 128 + tx * 16;
  const float* Grow = G + (size_t)bt * (12 * D_CH);
  float* Crow = Cbar + (size_t)bt * (12 * D_CH);

  #pragma unroll
  for (int r = 12; r < 16; ++r) gthird[r][t] = (_Float16)0.f;  // pad heads

  // ---- pass 1: scores[h][s] = sum_dd g[h][dd] * kv[s][dd] -----------------
  // stage i: c = i>>2 (dd third), c2 = (i>>1)&1 (s half), q2 = i&1 (dd half)
  if (w == 0) {
    tdm_stage_kv(xb + (size_t)0 * CH_STRIDE + 0,    &kvlds[0][0][0]); // i=0
    tdm_stage_kv(xb + (size_t)0 * CH_STRIDE + 1024, &kvlds[1][0][0]); // i=1
  }
  v8f acc0 = {}, acc1 = {};              // s = 16w (c2=0), s = 128+16w (c2=1)
  for (int i = 0; i < 12; ++i) {
    const int c = i >> 2, c2 = (i >> 1) & 1, q2 = i & 1, buf = i & 1;
    if ((i & 3) == 0) {                  // new dd third: stage g rows
      float gt[12];
      #pragma unroll
      for (int r = 0; r < 12; ++r) gt[r] = Grow[r * D_CH + c * 256 + t];
      #pragma unroll
      for (int r = 0; r < 12; ++r) gthird[r][t] = (_Float16)gt[r];
    }
    if (w == 0) {                        // stage i done (in-order TENSORcnt)
      if (i < 11) __builtin_amdgcn_s_wait_tensorcnt(1);
      else        __builtin_amdgcn_s_wait_tensorcnt(0);
    }
    __syncthreads();                     // kv + gthird visible
    const int sl = 16 * w + l16;         // this wave's score column (local s)
    #pragma unroll
    for (int kt = 0; kt < 4; ++kt) {
      int k0 = kt * 32;
      v8h alo = *(const v8h*)&gthird[l16][q2 * 128 + k0 + half * 8];
      v8h ahi = *(const v8h*)&gthird[l16][q2 * 128 + k0 + 16 + half * 8];
      v16h a = cat16(alo, ahi);
      const float* bp = &kvlds[buf][sl][k0 + half * 8];
      v4f f0 = *(const v4f*)(bp);
      v4f f1 = *(const v4f*)(bp + 4);
      v4f f2 = *(const v4f*)(bp + 16);
      v4f f3 = *(const v4f*)(bp + 20);
      v16h bb;
      #pragma unroll
      for (int e = 0; e < 4; ++e) {
        bb[e]      = (_Float16)f0[e];
        bb[4 + e]  = (_Float16)f1[e];
        bb[8 + e]  = (_Float16)f2[e];
        bb[12 + e] = (_Float16)f3[e];
      }
      if (c2 == 0) acc0 = wmma32(a, bb, acc0);
      else         acc1 = wmma32(a, bb, acc1);
    }
    __syncthreads();                     // readers of buf done
    if (w == 0 && i + 2 < 12) {          // refill freed buffer (overlaps i+1)
      int j = i + 2;
      int jc = j >> 2, jc2 = (j >> 1) & 1, jq2 = j & 1;
      tdm_stage_kv(xb + (size_t)(jc + 384 * jc2) * CH_STRIDE + jq2 * 1024,
                   &kvlds[j & 1][0][0]);
    }
  }
  // pre-issue first two cbar stages: DMA overlaps the softmax below.
  // cbar stage i: c = 2-(i>>2), q2 = (i>>1)&1, c2 = i&1
  if (w == 0) {
    tdm_stage_kv(xb + (size_t)(2 + 384 * 0) * CH_STRIDE + 0, &kvlds[0][0][0]);
    tdm_stage_kv(xb + (size_t)(2 + 384 * 1) * CH_STRIDE + 0, &kvlds[1][0][0]);
  }
  #pragma unroll
  for (int v = 0; v < 8; ++v) {          // C layout -> slds[h][s]
    int hh = v + 8 * half;
    slds[hh][16 * w + l16]       = acc0[v];
    slds[hh][128 + 16 * w + l16] = acc1[v];
  }
  __syncthreads();

  // ---- softmax over s (rows 2w, 2w+1 per wave) ----------------------------
  #pragma unroll
  for (int rr = 0; rr < 2; ++rr) {
    int r = 2 * w + rr;
    float vv[8], mx = -3.0e38f;
    #pragma unroll
    for (int i = 0; i < 8; ++i) { vv[i] = slds[r][lane + 32 * i]; mx = fmaxf(mx, vv[i]); }
    mx = wave_max(mx);
    float sm = 0.f;
    #pragma unroll
    for (int i = 0; i < 8; ++i) { vv[i] = __expf(vv[i] - mx); sm += vv[i]; }
    sm = wave_sum(sm);
    float inv = 1.f / sm;
    #pragma unroll
    for (int i = 0; i < 8; ++i) attnT[r][lane + 32 * i] = (_Float16)(vv[i] * inv);
  }

  // ---- pass 2: cbar[h][dd] = sum_s attn[h][s] * kv[s][dd] -----------------
  // thirds in reverse (c=2 staged last in pass 1 -> L2 temporal locality).
  v8f cacc = {};
  for (int i = 0; i < 12; ++i) {
    const int c = 2 - (i >> 2), q2 = (i >> 1) & 1, c2 = i & 1, buf = i & 1;
    if (w == 0) {
      if (i < 11) __builtin_amdgcn_s_wait_tensorcnt(1);
      else        __builtin_amdgcn_s_wait_tensorcnt(0);
    }
    __syncthreads();                     // kv ready; attnT visible (i=0)
    if (c2 == 0) cacc = v8f{};           // new (c,q2) output tile
    const int dd0 = 16 * w;              // this wave's dd tile (local in half)
    #pragma unroll
    for (int ks = 0; ks < 4; ++ks) {
      int s0 = ks * 32;
      int cs = c2 * 128;                 // global s offset for attn
      v8h alo = *(const v8h*)&attnT[l16][cs + s0 + half * 8];
      v8h ahi = *(const v8h*)&attnT[l16][cs + s0 + 16 + half * 8];
      v16h a = cat16(alo, ahi);
      v16h bb;                           // B = kv, k = s (strided column read)
      #pragma unroll
      for (int e = 0; e < 16; ++e)
        bb[e] = (_Float16)kvlds[buf][s0 + 8 * half + e + (e >= 8 ? 8 : 0)][dd0 + l16];
      cacc = wmma32(a, bb, cacc);
    }
    if (c2 == 1) {                       // (c,q2) tile complete -> slds
      #pragma unroll
      for (int v = 0; v < 8; ++v)
        slds[v + 8 * half][q2 * 128 + dd0 + l16] = cacc[v];
    }
    __syncthreads();                     // readers of buf (and slds) done
    if (w == 0 && i + 2 < 12) {
      int j = i + 2;
      int jc = 2 - (j >> 2), jq2 = (j >> 1) & 1, jc2 = j & 1;
      tdm_stage_kv(xb + (size_t)(jc + 384 * jc2) * CH_STRIDE + jq2 * 1024,
                   &kvlds[j & 1][0][0]);
    }
    if ((i & 3) == 3) {                  // finished a dd third: write it out
      float cb[12];
      #pragma unroll
      for (int r = 0; r < 12; ++r) cb[r] = slds[r][t];
      #pragma unroll
      for (int r = 0; r < 12; ++r) Crow[r * D_CH + c * 256 + t] = cb[r];
    }
  }
}

// ---------------------------------------------------------------------------
extern "C" void kernel_launch(void* const* d_in, const int* in_sizes, int n_in,
                              void* d_out, int out_size, void* d_ws, size_t ws_size,
                              hipStream_t stream) {
  const float* x         = (const float*)d_in[0];   // 8*768*128*128
  const float* in_proj_w = (const float*)d_in[1];   // 2304*768
  const float* in_proj_b = (const float*)d_in[2];   // 2304
  const float* out_w     = (const float*)d_in[3];   // 768*768
  const float* out_b     = (const float*)d_in[4];   // 768
  float* OUT = (float*)d_out;                       // 512*768 fp32

  // workspace layout (floats), total ~40.5 MB
  float* ws    = (float*)d_ws;
  float* Qmean = ws;                       // 512*768
  float* R     = ws + 393216;              // 512*768
  float* G     = ws + 786432;              // 512*9216
  float* Cbar  = ws + 5505024;             // 512*9216
  float* ctx   = ws + 10223616;            // 512*768

  const float* Wq = in_proj_w;
  const float* Wk = in_proj_w + 768 * 768;
  const float* Wv = in_proj_w + 2 * 768 * 768;
  const float* bq = in_proj_b;
  const float* bv = in_proj_b + 2 * 768;

  // 1) Qmean
  pool_kernel<<<NTILE, 256, 0, stream>>>(x, Qmean);

  // 2) R = Qmean @ Wq^T + bq          (M=512, N=768, K=768)
  gemm_f16w<<<dim3(12, 4, 1), 256, 0, stream>>>(
      Qmean, Wq, bq, R, 512, 768, 768, 768, 768, 768,
      /*transB=*/1, 1.0f, 0, 0, 0, 0);

  // 3) G_h = (R_h @ Wk_h) / 8         (batch 12: M=512, N=768, K=64)
  gemm_f16w<<<dim3(12, 4, 12), 256, 0, stream>>>(
      R, Wk, nullptr, G, 512, 768, 64, 768, 768, 9216,
      /*transB=*/0, 0.125f, 64, 64 * 768, 768, 0);

  // 4) attention: scores -> softmax -> Cbar   (double-buffered TDM pipeline)
  attn_kernel<<<NTILE, 256, 0, stream>>>(x, G, Cbar);

  // 5) ctx_h = Cbar_h @ Wv_h^T + bv_h (batch 12: M=512, N=64, K=768)
  gemm_f16w<<<dim3(1, 4, 12), 256, 0, stream>>>(
      Cbar, Wv, bv, ctx, 512, 64, 768, 9216, 768, 768,
      /*transB=*/1, 1.0f, 768, 64 * 768, 64, 64);

  // 6) OUT = ctx @ out_w^T + out_b    (M=512, N=768, K=768)
  gemm_f16w<<<dim3(12, 4, 1), 256, 0, stream>>>(
      ctx, out_w, out_b, OUT, 512, 768, 768, 768, 768, 768,
      /*transB=*/1, 1.0f, 0, 0, 0, 0);
}
